// GAT_22866405883989
// MI455X (gfx1250) — compile-verified, matrix operand order checked by reading, not run on hardware
//
#include <hip/hip_runtime.h>
#include <hip/hip_bf16.h>

typedef __attribute__((ext_vector_type(16))) __bf16 v16bf;
typedef __attribute__((ext_vector_type(8)))  float  v8f;

// ---------- helpers ----------
__device__ __forceinline__ unsigned short f2bf(float f) {
  unsigned u = __float_as_uint(f);
  u += 0x7FFFu + ((u >> 16) & 1u);           // round-to-nearest-even
  return (unsigned short)(u >> 16);
}
// order-preserving float<->uint map for atomic max on floats
__device__ __forceinline__ unsigned encf(float f) {
  unsigned u = __float_as_uint(f);
  return (u & 0x80000000u) ? ~u : (u | 0x80000000u);
}
__device__ __forceinline__ float decf(unsigned e) {
  unsigned u = (e & 0x80000000u) ? (e & 0x7FFFFFFFu) : ~e;
  return __uint_as_float(u);
}

// ---------- layer-0 projections: x[N,2] -> q|k|v|s planes [N,128] ----------
__global__ void k_proj0(const float* __restrict__ x,
                        const float* __restrict__ Wq, const float* __restrict__ bq,
                        const float* __restrict__ Wk, const float* __restrict__ bk,
                        const float* __restrict__ Wv, const float* __restrict__ bv,
                        const float* __restrict__ Ws, const float* __restrict__ bs,
                        float* __restrict__ qkvs, int N) {
  int n = blockIdx.x, c = threadIdx.x;
  float x0 = x[n * 2 + 0], x1 = x[n * 2 + 1];
  size_t P = (size_t)N * 128, i = (size_t)n * 128 + c;
  qkvs[0 * P + i] = x0 * Wq[c] + x1 * Wq[128 + c] + bq[c];
  qkvs[1 * P + i] = x0 * Wk[c] + x1 * Wk[128 + c] + bk[c];
  qkvs[2 * P + i] = x0 * Wv[c] + x1 * Wv[128 + c] + bv[c];
  qkvs[3 * P + i] = x0 * Ws[c] + x1 * Ws[128 + c] + bs[c];
}

// ---------- edge pass A: logits + segment max ----------
__global__ void k_edge_logits(const float* __restrict__ q, const float* __restrict__ k,
                              const float* __restrict__ We, const float* __restrict__ ea,
                              const int* __restrict__ src, const int* __restrict__ dst,
                              float* __restrict__ logits, unsigned* __restrict__ mu, int E) {
  __shared__ __align__(16) float sWe[128];
  if (threadIdx.x < 128) sWe[threadIdx.x] = We[threadIdx.x];
  __syncthreads();
  int idx = blockIdx.x * 256 + threadIdx.x;
  if (idx >= E * 2) return;
  int e = idx >> 1, h = idx & 1;
  int s = src[e], d = dst[e];
  float a = ea[e];
  const float4* qp = (const float4*)(q + (size_t)d * 128 + h * 64);
  const float4* kp = (const float4*)(k + (size_t)s * 128 + h * 64);
  const float4* wp = (const float4*)(sWe + h * 64);
  float acc = 0.f;
#pragma unroll
  for (int i = 0; i < 16; ++i) {
    float4 qq = qp[i], kk = kp[i], ww = wp[i];
    acc += qq.x * (kk.x + a * ww.x) + qq.y * (kk.y + a * ww.y)
         + qq.z * (kk.z + a * ww.z) + qq.w * (kk.w + a * ww.w);
  }
  float lg = acc * 0.125f;                   // 1/sqrt(64)
  logits[idx] = lg;
  atomicMax(&mu[d * 2 + h], encf(lg));
}

// ---------- edge pass B: exp-sum + weighted value scatter (1 wave / edge) ----------
__global__ void k_edge_accum(const float* __restrict__ v, const float* __restrict__ We,
                             const float* __restrict__ ea,
                             const int* __restrict__ src, const int* __restrict__ dst,
                             const float* __restrict__ logits, const unsigned* __restrict__ mu,
                             float* __restrict__ denom, float* __restrict__ num, int E) {
  __shared__ __align__(16) float sWe[128];
  if (threadIdx.x < 128) sWe[threadIdx.x] = We[threadIdx.x];
  __syncthreads();
  int e = blockIdx.x * 8 + (threadIdx.x >> 5);
  if (e >= E) return;
  int lane = threadIdx.x & 31;
  int c0 = lane * 4;
  int h = c0 >> 6;
  int s = src[e], d = dst[e];
  float a = ea[e];
  float ex = __expf(logits[e * 2 + h] - decf(mu[d * 2 + h]));
  float4 vv = *(const float4*)(v + (size_t)s * 128 + c0);
  float4 ww = *(const float4*)(sWe + c0);
  float* np = num + (size_t)d * 128 + c0;
  atomicAdd(np + 0, ex * (vv.x + a * ww.x));
  atomicAdd(np + 1, ex * (vv.y + a * ww.y));
  atomicAdd(np + 2, ex * (vv.z + a * ww.z));
  atomicAdd(np + 3, ex * (vv.w + a * ww.w));
  if ((c0 & 63) == 0) atomicAdd(&denom[d * 2 + h], ex);
}

// ---------- finalize layer0: normalize + skip + relu -> bf16 activations ----------
__global__ void k_finalize0(const float* __restrict__ num, const float* __restrict__ denom,
                            const float* __restrict__ s, unsigned short* __restrict__ hbf, int N) {
  int n = blockIdx.x, c = threadIdx.x;
  size_t i = (size_t)n * 128 + c;
  float val = num[i] / (denom[n * 2 + (c >> 6)] + 1e-16f) + s[i];
  hbf[i] = f2bf(fmaxf(val, 0.f));
}

// ---------- weight prep: fuse l1 {Wq,Wk,Wv,Ws} -> bf16 [512 col][128 k], bias cat ----------
__global__ void k_prep_wt(const float* __restrict__ Wq, const float* __restrict__ bq,
                          const float* __restrict__ Wk, const float* __restrict__ bk,
                          const float* __restrict__ Wv, const float* __restrict__ bv,
                          const float* __restrict__ Ws, const float* __restrict__ bs,
                          unsigned short* __restrict__ wt, float* __restrict__ bcat) {
  int i = blockIdx.x * 256 + threadIdx.x;
  if (i >= 512 * 128) return;
  int col = i >> 7, kk = i & 127;
  int plane = col >> 7;          // 0..3 : q,k,v,s
  int c = col & 127;
  const float* W = plane == 0 ? Wq : plane == 1 ? Wk : plane == 2 ? Wv : Ws;
  wt[col * 128 + kk] = f2bf(W[kk * 128 + c]);
  if (kk == 0) {
    const float* B = plane == 0 ? bq : plane == 1 ? bk : plane == 2 ? bv : bs;
    bcat[col] = B[c];
  }
}

// ---------- layer-1 projections via bf16 WMMA: [N,128] x [128,512] ----------
// one 16(M)x32(N) tile per wave: shared A fragment, two B fragments, 8 WMMAs.
// ISA 16-bit A layout: lanes 0-15 hold M=lane with K=0..7/16..23, lanes 16-31
// hold K=8..15/24..31; K pairs are contiguous -> dword loads, compiler merges
// consecutive pairs into global_load_b128 clauses.
__global__ void k_proj1_wmma(const unsigned short* __restrict__ hbf,
                             const unsigned short* __restrict__ wt,
                             const float* __restrict__ bcat,
                             float* __restrict__ qkvs, int N) {
  int lane = threadIdx.x & 31;
  int t = blockIdx.x * 8 + (threadIdx.x >> 5);
  int Mt = (N + 15) >> 4;
  if (t >= Mt * 16) return;                 // 16 double-wide N tiles (512/32)
  int tm = t >> 4, tn = (t & 15) * 2;       // tn in 16-col units, even
  int row   = tm * 16 + (lane & 15);
  int rowc  = row < N ? row : N - 1;
  bool full = (tm * 16 + 16) <= N;
  int col0  = tn * 16 + (lane & 15);
  int col1  = col0 + 16;
  int khi   = (lane >> 4) & 1;
  union { v16bf v; unsigned u[8]; } A, B0, B1;
  v8f acc0 = {}, acc1 = {};
  const unsigned short* hrow = hbf + (size_t)rowc * 128;
  const unsigned short* wc0  = wt + (size_t)col0 * 128;
  const unsigned short* wc1  = wt + (size_t)col1 * 128;
  for (int kb = 0; kb < 128; kb += 32) {
#pragma unroll
    for (int p = 0; p < 8; ++p) {
      // K of pair p: bit4 from p>=4, bit3 from lane-half, bits[2:1] from p&3
      int K0 = ((p >> 2) << 4) + (khi << 3) + ((p & 3) << 1);
      A.u[p]  = *(const unsigned*)(hrow + kb + K0);
      B0.u[p] = *(const unsigned*)(wc0 + kb + K0);
      B1.u[p] = *(const unsigned*)(wc1 + kb + K0);
    }
    acc0 = __builtin_amdgcn_wmma_f32_16x16x32_bf16(
        false, A.v, false, B0.v, (short)0, acc0, false, false);
    acc1 = __builtin_amdgcn_wmma_f32_16x16x32_bf16(
        false, A.v, false, B1.v, (short)0, acc1, false, false);
  }
  float bias0 = bcat[col0], bias1 = bcat[col1];
  int plane0 = col0 >> 7, cc0 = col0 & 127;
  int plane1 = col1 >> 7, cc1 = col1 & 127;
  float* o0 = qkvs + (size_t)plane0 * (size_t)N * 128;
  float* o1 = qkvs + (size_t)plane1 * (size_t)N * 128;
  int rbase = tm * 16 + 8 * khi;
  if (full) {                                // hot path: no per-row exec masking
#pragma unroll
    for (int r = 0; r < 8; ++r) {
      size_t rr = (size_t)(rbase + r) * 128;
      o0[rr + cc0] = acc0[r] + bias0;
      o1[rr + cc1] = acc1[r] + bias1;
    }
  } else {
#pragma unroll
    for (int r = 0; r < 8; ++r) {
      int ri = rbase + r;
      if (ri < N) {
        size_t rr = (size_t)ri * 128;
        o0[rr + cc0] = acc0[r] + bias0;
        o1[rr + cc1] = acc1[r] + bias1;
      }
    }
  }
}

// ---------- finalize layer1 + mean-pool accumulate ----------
__global__ void k_finalize1_pool(const float* __restrict__ num, const float* __restrict__ denom,
                                 const float* __restrict__ s, const int* __restrict__ batch,
                                 float* __restrict__ gsum, float* __restrict__ gcnt, int N) {
  int n = blockIdx.x, c = threadIdx.x;
  size_t i = (size_t)n * 128 + c;
  float val = num[i] / (denom[n * 2 + (c >> 6)] + 1e-16f) + s[i];
  val = fmaxf(val, 0.f);
  int b = batch[n];
  atomicAdd(&gsum[b * 128 + c], val);
  if (c == 0) atomicAdd(&gcnt[b], 1.0f);
}

// ---------- classifier MLP: g[64,128] -> relu(128) -> 30 ----------
__global__ void k_classifier(const float* __restrict__ gsum, const float* __restrict__ gcnt,
                             const float* __restrict__ W1, const float* __restrict__ b1,
                             const float* __restrict__ W2, const float* __restrict__ b2,
                             float* __restrict__ out) {
  __shared__ float sg[128], sh[128];
  int b = blockIdx.x, c = threadIdx.x;
  sg[c] = gsum[b * 128 + c] / fmaxf(gcnt[b], 1.0f);
  __syncthreads();
  float acc = b1[c];
  for (int k2 = 0; k2 < 128; ++k2) acc += sg[k2] * W1[k2 * 128 + c];
  sh[c] = fmaxf(acc, 0.f);
  __syncthreads();
  if (c < 30) {
    float a2 = b2[c];
    for (int k2 = 0; k2 < 128; ++k2) a2 += sh[k2] * W2[k2 * 30 + c];
    out[b * 30 + c] = a2;
  }
}

extern "C" void kernel_launch(void* const* d_in, const int* in_sizes, int n_in,
                              void* d_out, int out_size, void* d_ws, size_t ws_size,
                              hipStream_t stream) {
  (void)n_in; (void)out_size; (void)ws_size;
  const int N = in_sizes[0] / 2;     // 50000
  const int E = in_sizes[2];         // edge_attr is [E,1] -> 800000

  const float* x     = (const float*)d_in[0];
  const int*   eidx  = (const int*)  d_in[1];
  const float* eattr = (const float*)d_in[2];
  const int*   batch = (const int*)  d_in[3];
  const float* l0_Wq = (const float*)d_in[4];  const float* l0_bq = (const float*)d_in[5];
  const float* l0_Wk = (const float*)d_in[6];  const float* l0_bk = (const float*)d_in[7];
  const float* l0_Wv = (const float*)d_in[8];  const float* l0_bv = (const float*)d_in[9];
  const float* l0_We = (const float*)d_in[10];
  const float* l0_Ws = (const float*)d_in[11]; const float* l0_bs = (const float*)d_in[12];
  const float* l1_Wq = (const float*)d_in[13]; const float* l1_bq = (const float*)d_in[14];
  const float* l1_Wk = (const float*)d_in[15]; const float* l1_bk = (const float*)d_in[16];
  const float* l1_Wv = (const float*)d_in[17]; const float* l1_bv = (const float*)d_in[18];
  const float* l1_We = (const float*)d_in[19];
  const float* l1_Ws = (const float*)d_in[20]; const float* l1_bs = (const float*)d_in[21];
  const float* cW1   = (const float*)d_in[22]; const float* cb1   = (const float*)d_in[23];
  const float* cW2   = (const float*)d_in[24]; const float* cb2   = (const float*)d_in[25];

  // workspace carve-up
  char* ws = (char*)d_ws;
  size_t off = 0;
  auto carve = [&](size_t bytes) -> void* {
    void* p = ws + off;
    off = (off + bytes + 255) & ~(size_t)255;
    return p;
  };
  const size_t P = (size_t)N * 128;
  float*          qkvs   = (float*)         carve(4 * P * sizeof(float));   // q|k|v|s planes (reused both layers)
  float*          num    = (float*)         carve(P * sizeof(float));
  unsigned*       mu     = (unsigned*)      carve((size_t)N * 2 * sizeof(unsigned));
  float*          denom  = (float*)         carve((size_t)N * 2 * sizeof(float));
  float*          logits = (float*)         carve((size_t)E * 2 * sizeof(float));
  unsigned short* hbf    = (unsigned short*)carve(P * sizeof(unsigned short));
  unsigned short* wt     = (unsigned short*)carve(512 * 128 * sizeof(unsigned short));
  float*          bcat   = (float*)         carve(512 * sizeof(float));
  float*          gsum   = (float*)         carve(64 * 128 * sizeof(float));
  float*          gcnt   = (float*)         carve(64 * sizeof(float));

  const int* src = eidx;
  const int* dst = eidx + E;
  float* q = qkvs;          float* k = qkvs + P;
  float* v = qkvs + 2 * P;  float* s = qkvs + 3 * P;

  // zero accumulators (graph-capturable memset nodes)
  hipMemsetAsync(mu,    0, (size_t)N * 2 * sizeof(unsigned), stream);
  hipMemsetAsync(denom, 0, (size_t)N * 2 * sizeof(float), stream);
  hipMemsetAsync(num,   0, P * sizeof(float), stream);
  hipMemsetAsync(gsum,  0, 64 * 128 * sizeof(float), stream);
  hipMemsetAsync(gcnt,  0, 64 * sizeof(float), stream);

  // fused bf16 weight prep for layer-1 WMMA
  k_prep_wt<<<(512 * 128 + 255) / 256, 256, 0, stream>>>(
      l1_Wq, l1_bq, l1_Wk, l1_bk, l1_Wv, l1_bv, l1_Ws, l1_bs, wt, bcat);

  // ---- layer 0 ----
  k_proj0<<<N, 128, 0, stream>>>(x, l0_Wq, l0_bq, l0_Wk, l0_bk, l0_Wv, l0_bv,
                                 l0_Ws, l0_bs, qkvs, N);
  k_edge_logits<<<(E * 2 + 255) / 256, 256, 0, stream>>>(q, k, l0_We, eattr, src, dst,
                                                         logits, mu, E);
  k_edge_accum<<<(E + 7) / 8, 256, 0, stream>>>(v, l0_We, eattr, src, dst,
                                                logits, mu, denom, num, E);
  k_finalize0<<<N, 128, 0, stream>>>(num, denom, s, hbf, N);

  // reset accumulators for layer 1
  hipMemsetAsync(mu,    0, (size_t)N * 2 * sizeof(unsigned), stream);
  hipMemsetAsync(denom, 0, (size_t)N * 2 * sizeof(float), stream);
  hipMemsetAsync(num,   0, P * sizeof(float), stream);

  // ---- layer 1: WMMA projection then attention ----
  int Mt = (N + 15) >> 4;
  k_proj1_wmma<<<(Mt * 16 + 7) / 8, 256, 0, stream>>>(hbf, wt, bcat, qkvs, N);
  k_edge_logits<<<(E * 2 + 255) / 256, 256, 0, stream>>>(q, k, l1_We, eattr, src, dst,
                                                         logits, mu, E);
  k_edge_accum<<<(E + 7) / 8, 256, 0, stream>>>(v, l1_We, eattr, src, dst,
                                                logits, mu, denom, num, E);
  k_finalize1_pool<<<N, 128, 0, stream>>>(num, denom, s, batch, gsum, gcnt, N);

  // ---- classifier ----
  k_classifier<<<64, 128, 0, stream>>>(gsum, gcnt, cW1, cb1, cW2, cb2, (float*)d_out);
}
